// LinkPredictor_33964601377214
// MI455X (gfx1250) — matrix operand order repaired and checked by reading, not compiled, not run on hardware
//
#include <hip/hip_runtime.h>
#include <math.h>

#define N_NODES 50000
#define FEAT 128
#define NF (N_NODES * FEAT)

typedef __attribute__((ext_vector_type(16))) __bf16           v16bf;
typedef __attribute__((ext_vector_type(16))) unsigned short   v16u;
typedef __attribute__((ext_vector_type(8)))  unsigned short   v8u;
typedef __attribute__((ext_vector_type(8)))  float            v8f;

// float -> bf16 bits (round to nearest even)
__device__ __forceinline__ unsigned short f2bf(float f) {
    union { float f; unsigned int u; } c; c.f = f;
    unsigned int u = c.u;
    u += 0x7fffu + ((u >> 16) & 1u);
    return (unsigned short)(u >> 16);
}

// ---------------- degree / norm ----------------
__global__ void k_init_deg(float* __restrict__ deg) {
    int i = blockIdx.x * blockDim.x + threadIdx.x;
    if (i < N_NODES) deg[i] = 1.0f;            // self loop
}

__global__ void k_accum_deg(const int* __restrict__ dst, float* __restrict__ deg, int E) {
    int e = blockIdx.x * blockDim.x + threadIdx.x;
    if (e < E) atomicAdd(&deg[dst[e]], 1.0f);
}

__global__ void k_rsqrt_deg(float* __restrict__ deg) {
    int i = blockIdx.x * blockDim.x + threadIdx.x;
    if (i < N_NODES) deg[i] = rsqrtf(deg[i]);  // deg >= 1 always
}

// ---------------- bf16 packing (hoisted out of GEMM) ----------------
// pack 2 consecutive floats -> one u32 of 2 bf16
__global__ void k_pack_bf16(const float* __restrict__ X, unsigned int* __restrict__ P,
                            int npairs) {
    int i = blockIdx.x * blockDim.x + threadIdx.x;
    if (i < npairs) {
        const float2 t = ((const float2*)X)[i];
        P[i] = (unsigned int)f2bf(t.x) | ((unsigned int)f2bf(t.y) << 16);
    }
}

// W[k][n] (row-major, in_f x out) -> Wt[n][k] bf16 (column-major for B fragments)
__global__ void k_pack_wt(const float* __restrict__ W, unsigned short* __restrict__ Wt) {
    int i = blockIdx.x * blockDim.x + threadIdx.x;
    if (i < FEAT * FEAT) {
        int n = i >> 7, k = i & (FEAT - 1);    // write coalesced in k
        Wt[i] = f2bf(W[k * FEAT + n]);
    }
}

// ---------------- WMMA bf16 GEMM: H[N,128] = Xb[N,128] @ W[128,128] ----------------
// block = 256 threads = 8 waves; wave w owns 16x16 tile (rows blockIdx.x*16, cols w*16)
// Xb: row-major bf16; Wt: column-major bf16 (Wt[n][k])
__global__ void k_gemm_wmma(const unsigned short* __restrict__ Xb,
                            const unsigned short* __restrict__ Wt,
                            float* __restrict__ H) {
    const int wave = threadIdx.x >> 5;       // col tile 0..7
    const int lane = threadIdx.x & 31;
    const int half = lane >> 4;              // 0 or 1
    const int m16  = lane & 15;
    const int row_base = blockIdx.x * 16;
    const int col_base = wave * 16;
    const unsigned short* __restrict__ xr = Xb + (size_t)(row_base + m16) * FEAT;
    const unsigned short* __restrict__ wr = Wt + (size_t)(col_base + m16) * FEAT;

    v8f acc = {};
#pragma unroll
    for (int kb = 0; kb < FEAT; kb += 32) {
        // A fragment: lane holds K = kb+half*16 .. +15 of row (row_base+m16)
        v16u au = *(const v16u*)(xr + kb + half * 16);
        // B fragment: lane holds K = kb+half*16 .. +15 of column (col_base+m16)
        v16u bu = *(const v16u*)(wr + kb + half * 16);
        acc = __builtin_amdgcn_wmma_f32_16x16x32_bf16(
            false, __builtin_bit_cast(v16bf, au),
            false, __builtin_bit_cast(v16bf, bu),
            (short)0, acc, false, false);
    }
    // C/D layout: VGPR r -> M = r + half*8, N = m16
#pragma unroll
    for (int r = 0; r < 8; ++r)
        H[(size_t)(row_base + r + half * 8) * FEAT + col_base + m16] = acc[r];
}

// ---------------- aggregation ----------------
__global__ void k_selfloop(const float* __restrict__ h, const float* __restrict__ dinv,
                           float* __restrict__ out) {
    int idx = blockIdx.x * blockDim.x + threadIdx.x;
    if (idx < NF) {
        float d = dinv[idx >> 7];
        out[idx] = h[idx] * d * d;
    }
}

// one block (128 threads) per edge: out[dst] += h[src] * dinv[src]*dinv[dst]
__global__ void k_scatter(const float* __restrict__ h, const float* __restrict__ dinv,
                          const int* __restrict__ src, const int* __restrict__ dst,
                          float* __restrict__ out) {
    const int e = blockIdx.x;
    const int f = threadIdx.x;
    const int s = src[e], d = dst[e];
    const float w = dinv[s] * dinv[d];
    atomicAdd(&out[(size_t)d * FEAT + f], h[(size_t)s * FEAT + f] * w);
}

__global__ void k_bias_act(float* __restrict__ z, const float* __restrict__ b, int do_relu) {
    int idx = blockIdx.x * blockDim.x + threadIdx.x;
    if (idx < NF) {
        float v = z[idx] + b[idx & (FEAT - 1)];
        z[idx] = do_relu ? fmaxf(v, 0.0f) : v;
    }
}

// ---------------- decode: sigmoid(dot(z[ls], z[ld])) ----------------
__global__ void k_decode(const float* __restrict__ z, const int* __restrict__ ls,
                         const int* __restrict__ ld, float* __restrict__ out, int EL) {
    int gw   = (blockIdx.x * blockDim.x + threadIdx.x) >> 5;
    int lane = threadIdx.x & 31;
    if (gw >= EL) return;
    const float* __restrict__ za = z + (size_t)ls[gw] * FEAT;
    const float* __restrict__ zb = z + (size_t)ld[gw] * FEAT;
    float s = 0.0f;
#pragma unroll
    for (int f = lane; f < FEAT; f += 32) s += za[f] * zb[f];
#pragma unroll
    for (int off = 16; off > 0; off >>= 1) s += __shfl_down(s, off);
    if (lane == 0) out[gw] = 1.0f / (1.0f + __expf(-s));
}

extern "C" void kernel_launch(void* const* d_in, const int* in_sizes, int n_in,
                              void* d_out, int out_size, void* d_ws, size_t ws_size,
                              hipStream_t stream) {
    const float* x  = (const float*)d_in[0];
    const int*   ei = (const int*)d_in[1];   // [2, E]
    const int*   el = (const int*)d_in[2];   // [2, EL]
    const float* W1 = (const float*)d_in[3];
    const float* b1 = (const float*)d_in[4];
    const float* W2 = (const float*)d_in[5];
    const float* b2 = (const float*)d_in[6];
    float* out = (float*)d_out;

    const int E  = in_sizes[1] / 2;
    const int EL = in_sizes[2] / 2;
    const int* src = ei;
    const int* dst = ei + E;
    const int* ls  = el;
    const int* ld  = el + EL;

    float* dinv = (float*)d_ws;                               // N
    float* bufA = dinv + N_NODES;                             // NF (h)
    float* bufB = bufA + (size_t)NF;                          // NF (z)
    unsigned short* Xb = (unsigned short*)(bufB + (size_t)NF);// NF bf16
    unsigned short* Wt = Xb + (size_t)NF;                     // FEAT*FEAT bf16

    const int T    = 256;
    const int gN   = (N_NODES + T - 1) / T;
    const int gNF  = (NF + T - 1) / T;
    const int gNF2 = (NF / 2 + T - 1) / T;
    const int gE   = (E + T - 1) / T;
    const int gW   = (FEAT * FEAT + T - 1) / T;

    // degree + symmetric norm
    k_init_deg <<<gN, T, 0, stream>>>(dinv);
    k_accum_deg<<<gE, T, 0, stream>>>(dst, dinv, E);
    k_rsqrt_deg<<<gN, T, 0, stream>>>(dinv);

    // layer 1: h1 = x@W1 ; z1 = relu(agg(h1) + b1)
    k_pack_bf16<<<gNF2, T, 0, stream>>>(x, (unsigned int*)Xb, NF / 2);
    k_pack_wt  <<<gW, T, 0, stream>>>(W1, Wt);
    k_gemm_wmma<<<N_NODES / 16, 256, 0, stream>>>(Xb, Wt, bufA);
    k_selfloop <<<gNF, T, 0, stream>>>(bufA, dinv, bufB);
    k_scatter  <<<E, FEAT, 0, stream>>>(bufA, dinv, src, dst, bufB);
    k_bias_act <<<gNF, T, 0, stream>>>(bufB, b1, 1);

    // layer 2: h2 = z1@W2 ; z2 = agg(h2) + b2
    k_pack_bf16<<<gNF2, T, 0, stream>>>(bufB, (unsigned int*)Xb, NF / 2);
    k_pack_wt  <<<gW, T, 0, stream>>>(W2, Wt);
    k_gemm_wmma<<<N_NODES / 16, 256, 0, stream>>>(Xb, Wt, bufA);
    k_selfloop <<<gNF, T, 0, stream>>>(bufA, dinv, bufB);
    k_scatter  <<<E, FEAT, 0, stream>>>(bufA, dinv, src, dst, bufB);
    k_bias_act <<<gNF, T, 0, stream>>>(bufB, b2, 0);

    // decode
    k_decode<<<(EL + 7) / 8, 256, 0, stream>>>(bufB, ls, ld, out, EL);
}
